// MultiHeadAttention_62431644615193
// MI455X (gfx1250) — matrix operand order repaired and checked by reading, not compiled, not run on hardware
//
#include <hip/hip_runtime.h>
#include <hip/hip_bf16.h>

// ---------- types ----------
typedef __attribute__((ext_vector_type(16))) __bf16          bf16x16;
typedef __attribute__((ext_vector_type(8)))  float           f32x8;
typedef __attribute__((ext_vector_type(4)))  float           f32x4;
typedef __attribute__((ext_vector_type(8)))  unsigned short  u16x8;
typedef __attribute__((ext_vector_type(4)))  unsigned short  u16x4;
typedef __attribute__((ext_vector_type(4)))  unsigned int    u32x4;
typedef __attribute__((ext_vector_type(8)))  int             i32x8;
typedef __attribute__((ext_vector_type(4)))  int             i32x4;

#if __has_builtin(__builtin_amdgcn_tensor_load_to_lds) && __has_builtin(__builtin_amdgcn_s_wait_tensorcnt)
#define USE_TDM 1
#else
#define USE_TDM 0
#endif

// xor-lane swizzle within 16-lane halves: group-of-32 mode, and=0x1F, xor=off
#define SWZF(v, imm) __builtin_bit_cast(float, \
    __builtin_amdgcn_ds_swizzle(__builtin_bit_cast(int, (v)), (imm)))

// ---------- helpers ----------
__device__ __forceinline__ unsigned short f2bf(float f) {
    unsigned int u = __builtin_bit_cast(unsigned int, f);
    unsigned int r = u + 0x7FFFu + ((u >> 16) & 1u);   // round-to-nearest-even
    return (unsigned short)(r >> 16);
}
__device__ __forceinline__ float bf2f(unsigned short s) {
    unsigned int u = ((unsigned int)s) << 16;
    return __builtin_bit_cast(float, u);
}

// A-fragment (16x32 bf16, MxK): lane r<16 row=r, VGPR0-3 K=hh*8+0..7, VGPR4-7 K=16+hh*8+0..7
__device__ __forceinline__ bf16x16 frag_a_load(const unsigned short* p, int hh) {
    union { bf16x16 v; u16x8 h[2]; } u;
    u.h[0] = *(const u16x8*)(p + hh * 8);
    u.h[1] = *(const u16x8*)(p + 16 + hh * 8);
    return u.v;
}
// B-fragment (32x16 bf16, KxN): lane r holds col r; VGPR0-3 K=hh*8+0..7, VGPR4-7 K=16+hh*8+0..7
__device__ __forceinline__ bf16x16 frag_b_load(const unsigned short* p, int hh) {
    union { bf16x16 v; u16x8 h[2]; } u;
    u.h[0] = *(const u16x8*)(p + hh * 8);
    u.h[1] = *(const u16x8*)(p + 16 + hh * 8);
    return u.v;
}

#define BLKMN 128
#define KB 32

// ---------- generic WMMA GEMM: C[MxN] = A[MxK] * Bm[KxN], double-buffered LDS ----------
template<bool A_F32, bool C_BF16>
__global__ __launch_bounds__(256, 2) void gemm_wmma(
    const void* __restrict__ Ap, const float* __restrict__ Bm,
    void* __restrict__ Cp, int M, int N, int K)
{
    __shared__ __align__(16) unsigned short As[2][BLKMN * KB];   // [128][32] row-major
    __shared__ __align__(16) unsigned short Bs[2][BLKMN * KB];   // [n][k] transposed

    const int tid  = threadIdx.x;
    const int lane = tid & 31, wave = tid >> 5;
    const int wm = wave & 3, wn = wave >> 2;             // 4x2 wave grid: 32x64 per wave
    const int m0 = blockIdx.y * BLKMN, n0 = blockIdx.x * BLKMN;
    const int r = lane & 15, hh = lane >> 4;

    const float*          Af = (const float*)Ap;
    const unsigned short* Ab = (const unsigned short*)Ap;

    // Staging: load phase first (one clause, one wait), then store phase.
    auto stage = [&](int k0, int buf) {
        if constexpr (A_F32) {
            f32x4 a4[4];
            #pragma unroll
            for (int i = 0; i < 4; ++i) {                 // 1024 float4-chunks of A tile
                int c = tid + i * 256; int row = c >> 3; int col4 = (c & 7) * 4;
                a4[i] = *(const f32x4*)&Af[(size_t)(m0 + row) * K + k0 + col4];
            }
            f32x4 b4[4];
            #pragma unroll
            for (int i = 0; i < 4; ++i) {                 // 1024 float4-chunks of B tile
                int c = tid + i * 256; int kk = c >> 5; int nn4 = (c & 31) * 4;
                b4[i] = *(const f32x4*)&Bm[(size_t)(k0 + kk) * N + n0 + nn4];
            }
            #pragma unroll
            for (int i = 0; i < 4; ++i) {
                int c = tid + i * 256; int row = c >> 3; int col4 = (c & 7) * 4;
                u16x4 s4 = { f2bf(a4[i][0]), f2bf(a4[i][1]), f2bf(a4[i][2]), f2bf(a4[i][3]) };
                *(u16x4*)&As[buf][row * KB + col4] = s4;  // ds_store_b64
            }
            #pragma unroll
            for (int i = 0; i < 4; ++i) {
                int c = tid + i * 256; int kk = c >> 5; int nn4 = (c & 31) * 4;
                #pragma unroll
                for (int j = 0; j < 4; ++j)
                    Bs[buf][(nn4 + j) * KB + kk] = f2bf(b4[i][j]);   // transpose
            }
        } else {
            u16x8 a8[2];
            #pragma unroll
            for (int i = 0; i < 2; ++i) {                 // 512 b128-chunks of bf16 A tile
                int c = tid + i * 256; int row = c >> 2; int col8 = (c & 3) * 8;
                a8[i] = *(const u16x8*)&Ab[(size_t)(m0 + row) * K + k0 + col8];
            }
            f32x4 b4[4];
            #pragma unroll
            for (int i = 0; i < 4; ++i) {
                int c = tid + i * 256; int kk = c >> 5; int nn4 = (c & 31) * 4;
                b4[i] = *(const f32x4*)&Bm[(size_t)(k0 + kk) * N + n0 + nn4];
            }
            #pragma unroll
            for (int i = 0; i < 2; ++i) {
                int c = tid + i * 256; int row = c >> 2; int col8 = (c & 3) * 8;
                *(u16x8*)&As[buf][row * KB + col8] = a8[i];          // ds_store_b128
            }
            #pragma unroll
            for (int i = 0; i < 4; ++i) {
                int c = tid + i * 256; int kk = c >> 5; int nn4 = (c & 31) * 4;
                #pragma unroll
                for (int j = 0; j < 4; ++j)
                    Bs[buf][(nn4 + j) * KB + kk] = f2bf(b4[i][j]);
            }
        }
    };

    f32x8 acc[2][4] = {};

    stage(0, 0);
    __syncthreads();
    int buf = 0;
    for (int k0 = 0; k0 < K; k0 += KB) {
        if (k0 + KB < K) stage(k0 + KB, buf ^ 1);         // global loads overlap WMMA below
        if (k0 + 2 * KB < K)
            __builtin_prefetch(&Bm[(size_t)(k0 + 2 * KB) * N + n0 + tid], 0, 0);

        bf16x16 af[2], bfr[4];
        af[0] = frag_a_load(&As[buf][(wm * 32 +  0 + r) * KB], hh);
        af[1] = frag_a_load(&As[buf][(wm * 32 + 16 + r) * KB], hh);
        #pragma unroll
        for (int j = 0; j < 4; ++j)
            bfr[j] = frag_b_load(&Bs[buf][(wn * 64 + j * 16 + r) * KB], hh);
        #pragma unroll
        for (int i = 0; i < 2; ++i)
            #pragma unroll
            for (int j = 0; j < 4; ++j)
                acc[i][j] = __builtin_amdgcn_wmma_f32_16x16x32_bf16(
                    false, af[i], false, bfr[j], (short)0, acc[i][j], false, false);
        __syncthreads();
        buf ^= 1;
    }

    // epilogue: C layout lane<16 -> M=rr, lane>=16 -> M=rr+8, N=lane&15
    #pragma unroll
    for (int i = 0; i < 2; ++i)
        #pragma unroll
        for (int j = 0; j < 4; ++j)
            #pragma unroll
            for (int rr = 0; rr < 8; ++rr) {
                int m = m0 + wm * 32 + i * 16 + rr + 8 * hh;
                int n = n0 + wn * 64 + j * 16 + r;
                float val = acc[i][j][rr];
                if constexpr (C_BF16) ((unsigned short*)Cp)[(size_t)m * N + n] = f2bf(val);
                else                  ((float*)Cp)[(size_t)m * N + n] = val;
            }
}

// ---------- RoPE + head split:  qkv bf16 [B*S][3*D] -> q,k,v bf16 [B,H,S,64] ----------
#define HD 64
__global__ __launch_bounds__(256) void rope_split_kernel(
    const unsigned short* __restrict__ qkv,
    unsigned short* __restrict__ qo, unsigned short* __restrict__ ko,
    unsigned short* __restrict__ vo, int S, int H)
{
    const int D = H * HD;
    size_t t = (size_t)blockIdx.x * 256 + threadIdx.x;   // ((b*H+h)*S + s)*64 + d
    int d  = (int)(t & 63);
    int s  = (int)((t >> 6) % S);
    int bh = (int)(t / ((size_t)S * HD));
    int h  = bh % H;
    int b  = bh / H;

    size_t rowbase = ((size_t)b * S + s) * (size_t)(3 * D);
    int col  = h * HD + d;
    int colp = h * HD + (d ^ 1);

    // emb[d] = pos * inv_freq[d % 32]; inv_freq[i] = 10000^(-2i/64)
    float invf = __expf(-(float)(d & 31) * (9.210340371976184f / 32.0f));
    float ang = (float)s * invf;
    float c  = __cosf(ang);
    float sn = __sinf(ang);

    float xq  = bf2f(qkv[rowbase + 0 * D + col]);
    float xqp = bf2f(qkv[rowbase + 0 * D + colp]);
    float xk  = bf2f(qkv[rowbase + 1 * D + col]);
    float xkp = bf2f(qkv[rowbase + 1 * D + colp]);
    float rhq = (d & 1) ? xqp : -xqp;   // rotate_half interleaved
    float rhk = (d & 1) ? xkp : -xkp;

    qo[t] = f2bf(xq * c + rhq * sn);
    ko[t] = f2bf(xk * c + rhk * sn);
    vo[t] = qkv[rowbase + 2 * D + col];
}

// ---------- flash attention: Q,K,V bf16 [B,H,S,64] -> O bf16 [B*S][D] ----------
__global__ __launch_bounds__(256, 2) void attn_kernel(
    const unsigned short* __restrict__ Q, const unsigned short* __restrict__ Km,
    const unsigned short* __restrict__ V, unsigned short* __restrict__ O,
    int S, int H)
{
    __shared__ __align__(16) unsigned short VT[HD * 40];       // V transposed [dim][kv], stride 40
    __shared__ __align__(16) unsigned short Plds[8][16 * 32];  // per-wave P scratch
#if USE_TDM
    __shared__ __align__(16) unsigned short Ks[32 * HD];       // K tile [kv][dim] via TDM
#endif

    const int tid = threadIdx.x, lane = tid & 31, wave = tid >> 5;
    const int r = lane & 15, hh = lane >> 4;
    const int bh = blockIdx.y;
    const int h  = bh % H, b = bh / H;
    const int qblock = blockIdx.x * 128;
    const int qr0 = qblock + wave * 16;        // this wave's 16 query rows
    const size_t base = (size_t)bh * S * HD;
    const float scale = 0.125f;                // 1/sqrt(64)

    bf16x16 aq[2];
    {
        const unsigned short* qrow = Q + base + (size_t)(qr0 + r) * HD;
        aq[0] = frag_a_load(qrow,      hh);    // k-dim 0..31
        aq[1] = frag_a_load(qrow + 32, hh);    // k-dim 32..63
    }

    f32x8 acc[4] = {};
    float mrow[8], lrow[8];
    #pragma unroll
    for (int i = 0; i < 8; ++i) { mrow[i] = -3.0e38f; lrow[i] = 0.f; }

    const int kv_end = qblock + 128;           // causal: keys <= last query in block
    for (int kv0 = 0; kv0 < kv_end; kv0 += 32) {
#if USE_TDM
        // --- stage K tile [32][64] bf16 via Tensor Data Mover (wave 0 issues the DMA) ---
        if (wave == 0) {
            unsigned long long ga = (unsigned long long)(size_t)(Km + base + (size_t)kv0 * HD);
            unsigned lds_off = (unsigned)(size_t)(&Ks[0]);
            u32x4 g0 = { 1u,                                     // count=1, user mode
                         lds_off,                                // lds_addr
                         (unsigned)ga,                           // global_addr[31:0]
                         ((unsigned)(ga >> 32) & 0x01FFFFFFu) | 0x80000000u }; // addr[56:32] | type=2
            i32x8 g1 = { (int)0x00010000,                        // workgroup_mask=0, data_size=1 (2B)
                         (int)(64u << 16),                       // tensor_dim0=64 (low16)
                         (int)(((unsigned)S) << 16),             // dim0 hi=0 | tensor_dim1=S (low16)
                         (int)((64u << 16) | (((unsigned)S) >> 16)), // dim1 hi | tile_dim0=64
                         32,                                     // tile_dim1=32, tile_dim2=0
                         64,                                     // tensor_dim0_stride=64 (low32)
                         0, 0 };
            i32x4 z4 = {};
#if __clang_major__ >= 23
            i32x8 z8 = {};
            __builtin_amdgcn_tensor_load_to_lds(g0, g1, z4, z4, z8, 0);
#else
            __builtin_amdgcn_tensor_load_to_lds(g0, g1, z4, z4, 0);
#endif
        }
#endif
        // --- stage V tile transposed: one b128 row-chunk per thread, then scatter ---
        {
            int kvr = tid >> 3; int d0 = (tid & 7) * 8;
            u16x8 vv = *(const u16x8*)&V[base + (size_t)(kv0 + kvr) * HD + d0];
            #pragma unroll
            for (int j = 0; j < 8; ++j) VT[(d0 + j) * 40 + kvr] = vv[j];
        }
#if USE_TDM
        if (wave == 0) __builtin_amdgcn_s_wait_tensorcnt(0);   // after V loads issued
#endif
        __syncthreads();

        if (kv0 <= qr0 + 15) {                 // wave-uniform: EXEC all-1s at WMMA
            // ---- scores S = Q K^T (16x32, two 16x16 C tiles) ----
            f32x8 sc[2] = {};
            #pragma unroll
            for (int sub = 0; sub < 2; ++sub) {
#if USE_TDM
                const unsigned short* krow = &Ks[(sub * 16 + r) * HD];
#else
                const unsigned short* krow = Km + base + (size_t)(kv0 + sub * 16 + r) * HD;
#endif
                bf16x16 bk0 = frag_b_load(krow,      hh);
                bf16x16 bk1 = frag_b_load(krow + 32, hh);
                sc[sub] = __builtin_amdgcn_wmma_f32_16x16x32_bf16(
                    false, aq[0], false, bk0, (short)0, sc[sub], false, false);
                sc[sub] = __builtin_amdgcn_wmma_f32_16x16x32_bf16(
                    false, aq[1], false, bk1, (short)0, sc[sub], false, false);
            }

            // ---- online softmax (ds_swizzle xor-reductions within 16-lane halves) ----
            unsigned short* pl = &Plds[wave][0];
            float alpha[8];
            #pragma unroll
            for (int rr = 0; rr < 8; ++rr) {
                int row = qr0 + rr + 8 * hh;
                int c0 = kv0 + r, c1 = c0 + 16;
                float x0 = sc[0][rr] * scale; if (c0 > row) x0 = -3.0e38f;
                float x1 = sc[1][rr] * scale; if (c1 > row) x1 = -3.0e38f;
                float t = fmaxf(x0, x1);
                t = fmaxf(t, SWZF(t, 0x041F));
                t = fmaxf(t, SWZF(t, 0x081F));
                t = fmaxf(t, SWZF(t, 0x101F));
                t = fmaxf(t, SWZF(t, 0x201F));
                float mn = fmaxf(mrow[rr], t);
                float p0 = __expf(x0 - mn), p1 = __expf(x1 - mn);
                float sm = p0 + p1;
                sm += SWZF(sm, 0x041F);
                sm += SWZF(sm, 0x081F);
                sm += SWZF(sm, 0x101F);
                sm += SWZF(sm, 0x201F);
                alpha[rr] = __expf(mrow[rr] - mn);
                mrow[rr]  = mn;
                lrow[rr]  = lrow[rr] * alpha[rr] + sm;
                int prow = rr + 8 * hh;
                pl[prow * 32 + r]      = f2bf(p0);
                pl[prow * 32 + 16 + r] = f2bf(p1);
            }
            asm volatile("s_wait_dscnt 0" ::: "memory");   // P store -> A-frag reload
            bf16x16 pa = frag_a_load(pl + r * 32, hh);

            // ---- O += P V ----
            #pragma unroll
            for (int j = 0; j < 4; ++j) {
                #pragma unroll
                for (int rr = 0; rr < 8; ++rr) acc[j][rr] *= alpha[rr];
                bf16x16 bvj = frag_b_load(&VT[(j * 16 + r) * 40], hh);
                acc[j] = __builtin_amdgcn_wmma_f32_16x16x32_bf16(
                    false, pa, false, bvj, (short)0, acc[j], false, false);
            }
        }
        __syncthreads();
    }

    // epilogue -> [B*S][D] bf16, col = h*64 + j*16 + n
    const int D = H * HD;
    #pragma unroll
    for (int j = 0; j < 4; ++j)
        #pragma unroll
        for (int rr = 0; rr < 8; ++rr) {
            int row = qr0 + rr + 8 * hh;
            float val = acc[j][rr] / lrow[rr];
            O[((size_t)b * S + row) * D + h * HD + j * 16 + r] = f2bf(val);
        }
}

// ---------- launch ----------
extern "C" void kernel_launch(void* const* d_in, const int* in_sizes, int n_in,
                              void* d_out, int out_size, void* d_ws, size_t ws_size,
                              hipStream_t stream) {
    (void)in_sizes; (void)n_in; (void)out_size; (void)ws_size;
    const float* x     = (const float*)d_in[0];
    // d_in[1] = mask (causality handled analytically)
    const float* w_qkv = (const float*)d_in[2];
    const float* w_out = (const float*)d_in[3];
    float* out = (float*)d_out;

    const int Bsz = 2, S = 2048, D = 1024, H = 16;
    const int M = Bsz * S;                        // 4096

    // workspace: [0,24MB) qkv bf16 [4096][3072] (later reused as attn out [4096][1024])
    //            [24,32) q  [32,40) k  [40,48) v   (bf16 [B,H,S,64] each)
    unsigned short* qkv  = (unsigned short*)d_ws;
    unsigned short* attn = (unsigned short*)d_ws;
    unsigned short* qb = (unsigned short*)((char*)d_ws + (size_t)24 * 1024 * 1024);
    unsigned short* kb = qb + (size_t)M * D;
    unsigned short* vb = kb + (size_t)M * D;

    dim3 blk(256);
    // 1) qkv = x @ w_qkv   (fp32 in, bf16 out)
    gemm_wmma<true, true><<<dim3(3 * D / BLKMN, M / BLKMN), blk, 0, stream>>>(
        x, w_qkv, qkv, M, 3 * D, D);
    // 2) RoPE + split heads
    rope_split_kernel<<<dim3((unsigned)((size_t)M * D / 256)), blk, 0, stream>>>(
        qkv, qb, kb, vb, S, H);
    // 3) flash attention
    attn_kernel<<<dim3(S / 128, Bsz * H), blk, 0, stream>>>(qb, kb, vb, attn, S, H);
    // 4) out = attn @ w_out  (bf16 A, fp32 out)
    gemm_wmma<false, false><<<dim3(D / BLKMN, M / BLKMN), blk, 0, stream>>>(
        attn, w_out, out, M, D, D);
}
